// SFCSubConv_13408887898483
// MI455X (gfx1250) — matrix operand order repaired and checked by reading, not compiled, not run on hardware
//
#include <hip/hip_runtime.h>
#include <hip/hip_bf16.h>
#include <math.h>

typedef __attribute__((ext_vector_type(16))) _Float16 v16h;
typedef __attribute__((ext_vector_type(8)))  _Float16 v8h;
typedef __attribute__((ext_vector_type(8)))  float    v8f;

#define IN_DIM  128
#define OUT_DIM 128
#define LN_EPS  1e-5f
// prepacked W fragment table: 4 ksteps * 8 ntiles * 32 lanes * 16 f16
#define WPACK_ELEMS (4 * 8 * 32 * 16)

// ---------------------------------------------------------------------------
// Kernel: zero the degree accumulator
// ---------------------------------------------------------------------------
__global__ void zero_f32_kernel(float* __restrict__ p, int n) {
    int i = blockIdx.x * blockDim.x + threadIdx.x;
    if (i < n) p[i] = 0.0f;
}

// ---------------------------------------------------------------------------
// Kernel: weighted in-degree via float atomics (self-loop folded in later)
// ---------------------------------------------------------------------------
__global__ void degree_kernel(const long long* __restrict__ col,
                              const float* __restrict__ ew,
                              float* __restrict__ deg, int n_edges) {
    int e = blockIdx.x * blockDim.x + threadIdx.x;
    if (e < n_edges) atomicAdd(&deg[col[e]], ew[e]);
}

// ---------------------------------------------------------------------------
// Kernel: dinv[i] = rsqrt(deg[i] + 1)   (deg+1 >= 1, weights are in [0,1))
// ---------------------------------------------------------------------------
__global__ void dinv_kernel(const float* __restrict__ deg,
                            float* __restrict__ dinv, int n) {
    int i = blockIdx.x * blockDim.x + threadIdx.x;
    if (i < n) dinv[i] = rsqrtf(deg[i] + 1.0f);
}

// ---------------------------------------------------------------------------
// Kernel: split x into hi/lo f16 planes (Markidis split), row-major
// ---------------------------------------------------------------------------
__global__ void split_x_kernel(const float* __restrict__ x,
                               _Float16* __restrict__ xhi,
                               _Float16* __restrict__ xlo, long long total) {
    long long i = (long long)blockIdx.x * blockDim.x + threadIdx.x;
    if (i < total) {
        float f = x[i];
        _Float16 h = (_Float16)f;
        xhi[i] = h;
        xlo[i] = (_Float16)(f - (float)h);
    }
}

// ---------------------------------------------------------------------------
// Kernel: split W once into hi/lo f16, pre-swizzled into the exact
// 32x16 B-fragment lane layout so the GEMM loads each fragment as one
// contiguous 32-byte (2x b128) load per lane, zero conversion VALU.
// Index t = (((ks*8 + ntile)*32 + lane)*8 + v); element pair (2t, 2t+1)
// holds W[k][n], W[k+1][n] with k = ks*32 + (v>>2)*16 + (lane>>4)*8 + 2(v&3),
// n = ntile*16 + (lane&15).
// ---------------------------------------------------------------------------
__global__ void prep_w_kernel(const float* __restrict__ W,
                              _Float16* __restrict__ whi,
                              _Float16* __restrict__ wlo) {
    int t = blockIdx.x * blockDim.x + threadIdx.x;
    if (t >= 4 * 8 * 32 * 8) return;
    int v    = t & 7;
    int lane = (t >> 3) & 31;
    int nt   = (t >> 8) & 7;
    int ks   = t >> 11;
    int n = nt * 16 + (lane & 15);
    int k = ks * 32 + ((v & 4) << 2) + (lane >> 4) * 8 + ((v & 3) << 1);
    float f0 = W[(size_t)k * OUT_DIM + n];
    float f1 = W[(size_t)(k + 1) * OUT_DIM + n];
    _Float16 h0 = (_Float16)f0, h1 = (_Float16)f1;
    whi[2 * t]     = h0;
    whi[2 * t + 1] = h1;
    wlo[2 * t]     = (_Float16)(f0 - (float)h0);
    wlo[2 * t + 1] = (_Float16)(f1 - (float)h1);
}

// ---------------------------------------------------------------------------
// FAST GEMM: xw = x @ W from pre-split f16 planes + prepacked W fragments.
// Per wave-tile: 16 A b128 loads + 16 B b128 loads + 12 WMMAs, no cvt.
// A 16x32 f16 lane layout: lane half gives +8 on K; VGPRs 0-3 cover
// K[kb+half*8, +8), VGPRs 4-7 cover K[kb+16+half*8, +8) -> two contiguous
// 16-byte chunks per fragment.
// ---------------------------------------------------------------------------
__global__ __launch_bounds__(256) void gemm_wmma_fast_kernel(
        const _Float16* __restrict__ xhi, const _Float16* __restrict__ xlo,
        const _Float16* __restrict__ whi, const _Float16* __restrict__ wlo,
        float* __restrict__ xw, int n_nodes) {
    const int wave = threadIdx.x >> 5;        // 0..7 -> N tile
    const int lane = threadIdx.x & 31;
    const int half = lane >> 4;
    const int l16  = lane & 15;
    const int row0 = blockIdx.x * 16;
    const int n0   = wave * 16;

    int arow = row0 + l16;
    if (arow >= n_nodes) arow = n_nodes - 1;  // store below is guarded
    const _Float16* __restrict__ xh = xhi + (size_t)arow * IN_DIM;
    const _Float16* __restrict__ xl = xlo + (size_t)arow * IN_DIM;

    v8f c = {};
    #pragma unroll
    for (int ks = 0; ks < IN_DIM / 32; ++ks) {
        const int kb = ks * 32;
        v8h ah0 = *(const v8h*)(xh + kb + half * 8);
        v8h ah1 = *(const v8h*)(xh + kb + 16 + half * 8);
        v8h al0 = *(const v8h*)(xl + kb + half * 8);
        v8h al1 = *(const v8h*)(xl + kb + 16 + half * 8);
        v16h ah = __builtin_shufflevector(ah0, ah1, 0, 1, 2, 3, 4, 5, 6, 7,
                                          8, 9, 10, 11, 12, 13, 14, 15);
        v16h al = __builtin_shufflevector(al0, al1, 0, 1, 2, 3, 4, 5, 6, 7,
                                          8, 9, 10, 11, 12, 13, 14, 15);
        size_t boff = (((size_t)ks * 8 + wave) * 32 + lane) * 16;
        v16h bh = *(const v16h*)(whi + boff);
        v16h bl = *(const v16h*)(wlo + boff);
        c = __builtin_amdgcn_wmma_f32_16x16x32_f16(false, ah, false, bh,
                                                   (short)0, c, false, false);
        c = __builtin_amdgcn_wmma_f32_16x16x32_f16(false, al, false, bh,
                                                   (short)0, c, false, false);
        c = __builtin_amdgcn_wmma_f32_16x16x32_f16(false, ah, false, bl,
                                                   (short)0, c, false, false);
    }
    #pragma unroll
    for (int r = 0; r < 8; ++r) {
        int m = row0 + r + 8 * half;
        if (m < n_nodes) xw[(size_t)m * OUT_DIM + n0 + l16] = c[r];
    }
}

// ---------------------------------------------------------------------------
// FALLBACK GEMM (used only if workspace is too small for the split planes):
// inline fp32 -> hi/lo f16 split, same math. Proven to compile+WMMA.
// ---------------------------------------------------------------------------
__global__ __launch_bounds__(256) void gemm_wmma_inline_kernel(
        const float* __restrict__ x, const float* __restrict__ W,
        float* __restrict__ xw, int n_nodes) {
    const int wave = threadIdx.x >> 5;
    const int lane = threadIdx.x & 31;
    const int half = lane >> 4;
    const int l16  = lane & 15;
    const int row0 = blockIdx.x * 16;
    const int n0   = wave * 16;

    int arow = row0 + l16;
    if (arow >= n_nodes) arow = n_nodes - 1;
    const float* __restrict__ xrow = x + (size_t)arow * IN_DIM;

    v8f c = {};
    #pragma unroll
    for (int ks = 0; ks < IN_DIM / 32; ++ks) {
        const int kb = ks * 32;
        v16h ahi, alo, bhi, blo;
        #pragma unroll
        for (int v = 0; v < 8; ++v) {
            int k = kb + half * 16 + 2 * v;
            float f0 = xrow[k];
            float f1 = xrow[k + 1];
            _Float16 h0 = (_Float16)f0, h1 = (_Float16)f1;
            ahi[2 * v]     = h0;
            ahi[2 * v + 1] = h1;
            alo[2 * v]     = (_Float16)(f0 - (float)h0);
            alo[2 * v + 1] = (_Float16)(f1 - (float)h1);
        }
        #pragma unroll
        for (int v = 0; v < 8; ++v) {
            int k = kb + half * 16 + 2 * v;
            float f0 = W[(size_t)k       * OUT_DIM + n0 + l16];
            float f1 = W[(size_t)(k + 1) * OUT_DIM + n0 + l16];
            _Float16 h0 = (_Float16)f0, h1 = (_Float16)f1;
            bhi[2 * v]     = h0;
            bhi[2 * v + 1] = h1;
            blo[2 * v]     = (_Float16)(f0 - (float)h0);
            blo[2 * v + 1] = (_Float16)(f1 - (float)h1);
        }
        c = __builtin_amdgcn_wmma_f32_16x16x32_f16(false, ahi, false, bhi,
                                                   (short)0, c, false, false);
        c = __builtin_amdgcn_wmma_f32_16x16x32_f16(false, alo, false, bhi,
                                                   (short)0, c, false, false);
        c = __builtin_amdgcn_wmma_f32_16x16x32_f16(false, ahi, false, blo,
                                                   (short)0, c, false, false);
    }
    #pragma unroll
    for (int r = 0; r < 8; ++r) {
        int m = row0 + r + 8 * half;
        if (m < n_nodes) xw[(size_t)m * OUT_DIM + n0 + l16] = c[r];
    }
}

// ---------------------------------------------------------------------------
// Kernel: out[i][d] = dinv[i]^2 * xw[i][d] + b[d]   (self-loop + bias)
// ---------------------------------------------------------------------------
__global__ void init_out_kernel(const float* __restrict__ xw,
                                const float* __restrict__ dinv,
                                const float* __restrict__ b,
                                float* __restrict__ out, int n_nodes) {
    long long idx = (long long)blockIdx.x * blockDim.x + threadIdx.x;
    long long total = (long long)n_nodes * OUT_DIM;
    if (idx < total) {
        long long node = idx >> 7;
        int d = (int)(idx & (OUT_DIM - 1));
        float di = dinv[node];
        out[idx] = di * di * xw[idx] + b[d];
    }
}

// ---------------------------------------------------------------------------
// Kernel: edge scatter: out[col] += dinv[row]*w*dinv[col] * xw[row]
// xw + out (~103 MB) are L2-resident (192 MB L2) -> atomics run at L2 rate.
// ---------------------------------------------------------------------------
__global__ __launch_bounds__(256) void scatter_kernel(
        const long long* __restrict__ row, const long long* __restrict__ col,
        const float* __restrict__ ew, const float* __restrict__ dinv,
        const float* __restrict__ xw, float* __restrict__ out, int n_edges) {
    int e = blockIdx.x * 2 + (threadIdx.x >> 7);
    int d = threadIdx.x & (OUT_DIM - 1);
    if (e >= n_edges) return;
    long long r = row[e];
    long long cdst = col[e];
    float norm = dinv[r] * ew[e] * dinv[cdst];
    atomicAdd(&out[cdst * OUT_DIM + d], norm * xw[r * OUT_DIM + d]);
}

// ---------------------------------------------------------------------------
// Kernel: LayerNorm over feature dim, one wave32 per node
// ---------------------------------------------------------------------------
__global__ __launch_bounds__(256) void layernorm_kernel(
        float* __restrict__ out, const float* __restrict__ gamma,
        const float* __restrict__ beta, int n_nodes) {
    int node = blockIdx.x * (blockDim.x >> 5) + (threadIdx.x >> 5);
    int lane = threadIdx.x & 31;
    if (node >= n_nodes) return;
    float* p = out + (size_t)node * OUT_DIM;
    float v[4];
    float s = 0.0f;
    #pragma unroll
    for (int i = 0; i < 4; ++i) { v[i] = p[lane + 32 * i]; s += v[i]; }
    #pragma unroll
    for (int off = 16; off > 0; off >>= 1) s += __shfl_xor(s, off, 32);
    float mu = s * (1.0f / OUT_DIM);
    float s2 = 0.0f;
    #pragma unroll
    for (int i = 0; i < 4; ++i) { float dd = v[i] - mu; s2 += dd * dd; }
    #pragma unroll
    for (int off = 16; off > 0; off >>= 1) s2 += __shfl_xor(s2, off, 32);
    float inv = rsqrtf(s2 * (1.0f / OUT_DIM) + LN_EPS);
    #pragma unroll
    for (int i = 0; i < 4; ++i) {
        int d = lane + 32 * i;
        p[d] = (v[i] - mu) * inv * gamma[d] + beta[d];
    }
}

// ---------------------------------------------------------------------------
static inline size_t align_up(size_t v, size_t a) { return (v + a - 1) & ~(a - 1); }

extern "C" void kernel_launch(void* const* d_in, const int* in_sizes, int n_in,
                              void* d_out, int out_size, void* d_ws, size_t ws_size,
                              hipStream_t stream) {
    const float*     x     = (const float*)d_in[0];
    const long long* ei    = (const long long*)d_in[1];   // int64 [2, E]
    const float*     ew    = (const float*)d_in[2];
    const float*     W     = (const float*)d_in[3];
    const float*     b     = (const float*)d_in[4];
    const float*     gamma = (const float*)d_in[5];
    const float*     beta  = (const float*)d_in[6];
    float*           out   = (float*)d_out;

    const int N = in_sizes[0] / IN_DIM;   // 100000
    const int E = in_sizes[2];            // 400000
    const long long* row = ei;            // edge_index[0] = source
    const long long* col = ei + E;        // edge_index[1] = destination
    const long long  NF  = (long long)N * IN_DIM;

    // workspace layout (byte offsets, 64B aligned):
    //   deg[N] f32 | dinv[N] f32 | xw[N*128] f32 | whi/wlo (64KB) | xhi/xlo f16
    char* ws = (char*)d_ws;
    size_t off = 0;
    float* deg  = (float*)(ws + off); off = align_up(off + (size_t)N * 4, 64);
    float* dinv = (float*)(ws + off); off = align_up(off + (size_t)N * 4, 64);
    float* xw   = (float*)(ws + off); off = align_up(off + (size_t)NF * 4, 64);
    size_t off_w = off;
    _Float16* whi = (_Float16*)(ws + off); off = align_up(off + WPACK_ELEMS * 2, 64);
    _Float16* wlo = (_Float16*)(ws + off); off = align_up(off + WPACK_ELEMS * 2, 64);
    _Float16* xhi = (_Float16*)(ws + off); off = align_up(off + (size_t)NF * 2, 64);
    _Float16* xlo = (_Float16*)(ws + off); off = align_up(off + (size_t)NF * 2, 64);
    const bool fast = (off <= ws_size);
    (void)off_w;

    zero_f32_kernel<<<(N + 255) / 256, 256, 0, stream>>>(deg, N);
    degree_kernel<<<(E + 255) / 256, 256, 0, stream>>>(col, ew, deg, E);
    dinv_kernel<<<(N + 255) / 256, 256, 0, stream>>>(deg, dinv, N);

    if (fast) {
        prep_w_kernel<<<(4 * 8 * 32 * 8 + 255) / 256, 256, 0, stream>>>(W, whi, wlo);
        split_x_kernel<<<(unsigned)((NF + 255) / 256), 256, 0, stream>>>(x, xhi, xlo, NF);
        gemm_wmma_fast_kernel<<<(N + 15) / 16, 256, 0, stream>>>(xhi, xlo, whi, wlo, xw, N);
    } else {
        gemm_wmma_inline_kernel<<<(N + 15) / 16, 256, 0, stream>>>(x, W, xw, N);
    }

    {
        long long total = NF;
        init_out_kernel<<<(unsigned)((total + 255) / 256), 256, 0, stream>>>(
            xw, dinv, b, out, N);
    }
    scatter_kernel<<<(E + 1) / 2, 256, 0, stream>>>(row, col, ew, dinv, xw, out, E);
    layernorm_kernel<<<(N + 7) / 8, 256, 0, stream>>>(out, gamma, beta, N);
}
